// FeatDecoder_29807073034858
// MI455X (gfx1250) — matrix operand order repaired and checked by reading, not compiled, not run on hardware
//
#include <hip/hip_runtime.h>
#include <hip/hip_bf16.h>

// ---------------------------------------------------------------------------
// GRU feature decoder for MI455X (gfx1250), wave32 + WMMA bf16.
//
// Roofline: recurrent GEMM h@W_hh.T dominates (~830 GFLOP). Everything else
// (h0, z_in, gi_base precompute ~8 GFLOP; <100MB HBM traffic at 23.3TB/s) is
// noise, so the recurrent kernel keeps h resident on-chip and drives
// v_wmma_f32_16x16x32_bf16 from LDS (A = h tile) and L2 (B = packed W_hh).
//
// Round-1 fix: B-fragments are gathered into a 6-element register array
// before the WMMA group so all 12 global_load_b128 are in flight at once
// (distinct dest VGPRs -> partial s_wait_loadcnt instead of 6 serialized
// full waits per k-chunk).
// ---------------------------------------------------------------------------

#define BS      4096
#define Z_DIM   512
#define HID     1024
#define Z_IN    128
#define N_STEP  32
#define G3      3072          // 3*HID
#define ROWS    16            // batch rows per workgroup (one WMMA M tile)
#define S16     1032          // padded bf16 LDS row stride (bank spread)

typedef __bf16 v8bf  __attribute__((ext_vector_type(8)));
typedef __bf16 v16bf __attribute__((ext_vector_type(16)));
typedef float  v8f   __attribute__((ext_vector_type(8)));

__device__ __forceinline__ float sigm(float x)   { return 1.0f / (1.0f + __expf(-x)); }
__device__ __forceinline__ float tanh_f(float x) { return 2.0f / (1.0f + __expf(-2.0f * x)) - 1.0f; }

// ---------------------------------------------------------------------------
// Prep 1: h0 = z @ W_zh.T + b_zh         [4096,1024]
// ---------------------------------------------------------------------------
__global__ void k_h0(const float* __restrict__ z, const float* __restrict__ Wzh,
                     const float* __restrict__ bzh, float* __restrict__ h0) {
  int idx = blockIdx.x * 256 + threadIdx.x;          // 4096*1024 threads
  int row = idx >> 10, col = idx & 1023;
  const float4* zr = (const float4*)(z   + (size_t)row * Z_DIM);
  const float4* wr = (const float4*)(Wzh + (size_t)col * Z_DIM);
  float s = 0.f;
#pragma unroll 4
  for (int i = 0; i < Z_DIM / 4; ++i) {
    float4 a = zr[i], b = wr[i];
    s += a.x * b.x + a.y * b.y + a.z * b.z + a.w * b.w;
  }
  h0[idx] = s + bzh[col];
}

// ---------------------------------------------------------------------------
// Prep 2: z_in = z @ W_zi.T + b_zi       [4096,128]
// ---------------------------------------------------------------------------
__global__ void k_zin(const float* __restrict__ z, const float* __restrict__ Wzi,
                      const float* __restrict__ bzi, float* __restrict__ zin) {
  int idx = blockIdx.x * 256 + threadIdx.x;          // 4096*128 threads
  int row = idx >> 7, col = idx & 127;
  const float4* zr = (const float4*)(z   + (size_t)row * Z_DIM);
  const float4* wr = (const float4*)(Wzi + (size_t)col * Z_DIM);
  float s = 0.f;
#pragma unroll 4
  for (int i = 0; i < Z_DIM / 4; ++i) {
    float4 a = zr[i], b = wr[i];
    s += a.x * b.x + a.y * b.y + a.z * b.z + a.w * b.w;
  }
  zin[idx] = s + bzi[col];
}

// ---------------------------------------------------------------------------
// Prep 3: gi_base = z_in @ W_ih[:,3:].T + b_ih     [4096,3072]
// (the z_in part of gi is step-invariant; token part is rank-3, done per step)
// ---------------------------------------------------------------------------
__global__ void k_gib(const float* __restrict__ zin, const float* __restrict__ Wih,
                      const float* __restrict__ bih, float* __restrict__ gib) {
  int idx = blockIdx.x * 256 + threadIdx.x;          // 4096*3072 threads
  int row = idx / G3, col = idx % G3;
  const float* zr = zin + (size_t)row * Z_IN;
  const float* wr = Wih + (size_t)col * 131 + 3;     // skip 3 token columns
  float s = bih[col];
#pragma unroll 8
  for (int i = 0; i < Z_IN; ++i) s += zr[i] * wr[i];
  gib[idx] = s;
}

// ---------------------------------------------------------------------------
// Prep 4: pack W_hh [3072,1024] f32 -> bf16 in WMMA B-operand lane order.
// B (32x16 KxN, bf16): lane L holds n = L%16, element j (0..15) holds
// k = (L/16)*16 + j. Flat: Bpack[((ntile*32 + kchunk)*32 + lane)*16 + j].
// Each lane then fetches one 32B-aligned v16bf = 2x global_load_b128.
// ---------------------------------------------------------------------------
__global__ void k_pack(const float* __restrict__ Whh, __bf16* __restrict__ Bpack) {
  int idx  = blockIdx.x * 256 + threadIdx.x;         // 3072*1024 threads
  int j    = idx & 15;
  int lane = (idx >> 4) & 31;
  int kc   = (idx >> 9) & 31;
  int nt   = idx >> 14;
  int n = nt * 16 + (lane & 15);
  int k = kc * 32 + (lane >> 4) * 16 + j;
  Bpack[idx] = (__bf16)Whh[(size_t)n * HID + k];
}

// ---------------------------------------------------------------------------
// Recurrent kernel: 256 blocks x 1024 threads (32 waves), ROWS=16 rows/block,
// persistent over 32 GRU steps. Wave w owns cols [32w,32w+32) of each gate.
// ---------------------------------------------------------------------------
__global__ void k_gru(const float*  __restrict__ h0,
                      const float*  __restrict__ gi_base,
                      const __bf16* __restrict__ Bpack,
                      const float*  __restrict__ b_hh,
                      const float*  __restrict__ W_ih,
                      const float*  __restrict__ W_out,
                      const float*  __restrict__ b_out,
                      const float*  __restrict__ init_in,
                      float*        __restrict__ out) {
  extern __shared__ char smem[];
  __bf16* h16 = (__bf16*)smem;                               // 16*1032*2 = 33024 B
  float*  h32 = (float*)(smem + ROWS * S16 * 2);             // 16*1024*4 = 65536 B
  float*  tok = (float*)(smem + ROWS * S16 * 2 + ROWS * HID * 4); // 16*3*4 = 192 B

  const int tid  = threadIdx.x;
  const int lane = tid & 31;
  const int w    = tid >> 5;          // wave id 0..31
  const int kg   = lane >> 4;         // half-wave selector
  const int ln16 = lane & 15;
  const int rowbase = blockIdx.x * ROWS;

  // ---- init: h state (regs: wave-owned fp32 cols; LDS: bf16 + fp32 full tile)
  float hreg[2][8];
#pragma unroll
  for (int nt = 0; nt < 2; ++nt) {
    int col = w * 32 + nt * 16 + ln16;
#pragma unroll
    for (int v = 0; v < 8; ++v) {
      int m = v + 8 * kg;                       // C/D layout: VGPR v -> row v (+8)
      float hv = h0[(size_t)(rowbase + m) * HID + col];
      hreg[nt][v] = hv;
      h16[m * S16 + col] = (__bf16)hv;
      h32[m * HID + col] = hv;
    }
  }
  if (tid < ROWS * 3) tok[tid] = init_in[tid % 3];

  // wave's six 16-col N tiles: gate g, sub-tile nt -> global tile g*64 + w*2 + nt
  int ntile[6];
#pragma unroll
  for (int g = 0; g < 3; ++g)
#pragma unroll
    for (int nt = 0; nt < 2; ++nt) ntile[g * 2 + nt] = g * 64 + w * 2 + nt;

  __syncthreads();

  for (int step = 0; step < N_STEP; ++step) {
    // ============== gh = h @ W_hh.T  (wave's 96 cols) via WMMA bf16 ==========
    v8f acc[6];
#pragma unroll
    for (int t = 0; t < 6; ++t)
#pragma unroll
      for (int e = 0; e < 8; ++e) acc[t][e] = 0.0f;

#pragma unroll 2
    for (int kc = 0; kc < 32; ++kc) {
      // A frag (16x32 MxK bf16): lane m=ln16; elems 0..15 k=kc*32+kg*16+j,
      // contiguous -> two 16B LDS loads
      int abase = ln16 * S16 + kc * 32 + kg * 16;
      v8bf alo = *(const v8bf*)(h16 + abase);
      v8bf ahi = *(const v8bf*)(h16 + abase + 8);
      v16bf a = __builtin_shufflevector(alo, ahi, 0, 1, 2, 3, 4, 5, 6, 7,
                                        8, 9, 10, 11, 12, 13, 14, 15);

      // Gather all six B fragments into distinct registers first so the
      // twelve global_load_b128 are simultaneously in flight (partial
      // s_wait_loadcnt before each WMMA instead of full round-trips).
      v16bf bf[6];
#pragma unroll
      for (int t = 0; t < 6; ++t)
        bf[t] = *(const v16bf*)(Bpack +
                                ((size_t)(ntile[t] * 32 + kc) * 32 + lane) * 16);
#pragma unroll
      for (int t = 0; t < 6; ++t)
        acc[t] = __builtin_amdgcn_wmma_f32_16x16x32_bf16(
            false, a, false, bf[t], (short)0, acc[t], false, false);
    }
    __syncthreads();   // all waves done reading h16 of this step

    // ============== gate math for wave-owned columns ========================
    float w3v[6][3], bh[6];
#pragma unroll
    for (int t = 0; t < 6; ++t) {
      int g = t >> 1, nt = t & 1;
      int n = g * HID + w * 32 + nt * 16 + ln16;
      w3v[t][0] = W_ih[(size_t)n * 131 + 0];
      w3v[t][1] = W_ih[(size_t)n * 131 + 1];
      w3v[t][2] = W_ih[(size_t)n * 131 + 2];
      bh[t] = b_hh[n];
    }

#pragma unroll
    for (int v = 0; v < 8; ++v) {
      int m = v + 8 * kg;
      float t0 = tok[m * 3 + 0], t1 = tok[m * 3 + 1], t2 = tok[m * 3 + 2];
      const float* gr = gi_base + (size_t)(rowbase + m) * G3;
#pragma unroll
      for (int nt = 0; nt < 2; ++nt) {
        int col = w * 32 + nt * 16 + ln16;
        float gi_r = gr[col]           + t0 * w3v[nt][0]     + t1 * w3v[nt][1]     + t2 * w3v[nt][2];
        float gi_u = gr[HID + col]     + t0 * w3v[2 + nt][0] + t1 * w3v[2 + nt][1] + t2 * w3v[2 + nt][2];
        float gi_n = gr[2 * HID + col] + t0 * w3v[4 + nt][0] + t1 * w3v[4 + nt][1] + t2 * w3v[4 + nt][2];
        float rr = sigm(gi_r + acc[nt][v] + bh[nt]);
        float uu = sigm(gi_u + acc[2 + nt][v] + bh[2 + nt]);
        float nn = tanh_f(gi_n + rr * (acc[4 + nt][v] + bh[4 + nt]));
        float hn = (1.0f - uu) * nn + uu * hreg[nt][v];
        hreg[nt][v] = hn;
        h16[m * S16 + col] = (__bf16)hn;
        h32[m * HID + col] = hn;
      }
    }
    __syncthreads();   // h_new visible to all waves

    // ============== o = h_new @ W_out.T + b_out; emit & next token ==========
    if (w < ROWS) {
      float s0 = 0.f, s1 = 0.f, s2 = 0.f;
#pragma unroll 4
      for (int i = 0; i < HID / 32; ++i) {
        int k = i * 32 + lane;
        float hv = h32[w * HID + k];
        s0 += hv * W_out[k];
        s1 += hv * W_out[HID + k];
        s2 += hv * W_out[2 * HID + k];
      }
#pragma unroll
      for (int off = 16; off > 0; off >>= 1) {
        s0 += __shfl_xor(s0, off, 32);
        s1 += __shfl_xor(s1, off, 32);
        s2 += __shfl_xor(s2, off, 32);
      }
      if (lane == 0) {
        float bass = sigm(s0 + b_out[0]);
        float ri   = s1 + b_out[1];
        float rhy  = sigm(s2 + b_out[2]);
        size_t ob = ((size_t)(rowbase + w) * N_STEP + step) * 3;
        out[ob + 0] = bass;
        out[ob + 1] = ri;
        out[ob + 2] = rhy;
        tok[w * 3 + 0] = bass;               // next token (read after next
        tok[w * 3 + 1] = ri;                 //  step's post-GEMM barrier)
        tok[w * 3 + 2] = (rhy > 0.5f) ? 1.0f : 0.0f;
      }
    }
    // no barrier needed here: next step's post-GEMM __syncthreads orders the
    // token/h32 writes above against their next readers.
  }
}

// ---------------------------------------------------------------------------
extern "C" void kernel_launch(void* const* d_in, const int* in_sizes, int n_in,
                              void* d_out, int out_size, void* d_ws, size_t ws_size,
                              hipStream_t stream) {
  const float* z     = (const float*)d_in[0];
  const float* W_zh  = (const float*)d_in[3];
  const float* b_zh  = (const float*)d_in[4];
  const float* W_zi  = (const float*)d_in[5];
  const float* b_zi  = (const float*)d_in[6];
  const float* W_ih  = (const float*)d_in[7];
  const float* b_ih  = (const float*)d_in[8];
  const float* W_hh  = (const float*)d_in[9];
  const float* b_hh  = (const float*)d_in[10];
  const float* W_out = (const float*)d_in[11];
  const float* b_out = (const float*)d_in[12];
  const float* initi = (const float*)d_in[13];
  float* out = (float*)d_out;

  // workspace layout
  char* ws = (char*)d_ws;
  float*  h0    = (float*)(ws);                              // 16.78 MB
  float*  zin   = (float*)(ws + (size_t)16777216);           //  2.10 MB
  float*  gib   = (float*)(ws + (size_t)18874368);           // 50.33 MB
  __bf16* Bpack = (__bf16*)(ws + (size_t)69206016);          //  6.29 MB

  k_h0  <<<dim3((BS * HID)   / 256), dim3(256), 0, stream>>>(z, W_zh, b_zh, h0);
  k_zin <<<dim3((BS * Z_IN)  / 256), dim3(256), 0, stream>>>(z, W_zi, b_zi, zin);
  k_gib <<<dim3((BS * G3)    / 256), dim3(256), 0, stream>>>(zin, W_ih, b_ih, gib);
  k_pack<<<dim3((G3 * HID)   / 256), dim3(256), 0, stream>>>(W_hh, Bpack);

  const int smem = ROWS * S16 * 2 + ROWS * HID * 4 + ROWS * 3 * 4; // 98752 B
  k_gru<<<dim3(BS / ROWS), dim3(1024), smem, stream>>>(
      h0, gib, Bpack, b_hh, W_ih, W_out, b_out, initi, out);
}
